// MoeExperts_31928786879171
// MI455X (gfx1250) — compile-verified, matrix operand order checked by reading, not run on hardware
//
#include <hip/hip_runtime.h>
#include <stdint.h>

// ---------------------------------------------------------------------------
// MoE experts (top-2 of 8), T=4096, H=2048, I=5632, fp32 in/out.
// Strategy: device-side token gather per expert -> bf16 WMMA GEMMs
// (fp32 weights converted to bf16 while staging to LDS), SiLU fused into
// gate_up epilogue, weighted atomic scatter in down epilogue.
// CDNA5 paths: v_wmma_f32_16x16x32_bf16 for the math, TENSOR_LOAD_TO_LDS (TDM)
// for the regular-strided activation tile in the down GEMM.
// ---------------------------------------------------------------------------

typedef __attribute__((ext_vector_type(16))) __bf16 v16bf;
typedef __attribute__((ext_vector_type(8)))  float  v8f;
typedef __attribute__((ext_vector_type(4)))  unsigned int v4u;
typedef __attribute__((ext_vector_type(8)))  int v8i;
typedef __attribute__((ext_vector_type(4)))  int v4i;

constexpr int NE   = 8;
constexpr int HID  = 2048;
constexpr int INTR = 5632;
constexpr int NTOK = 4096;

constexpr int TM   = 128;   // rows per block tile
constexpr int TN   = 64;    // cols per block tile
constexpr int TK   = 32;    // K step (bf16 wmma K)
constexpr int LSTR = 48;    // LDS row stride in elements (32 + 16 pad, keeps 16B align)
constexpr int MAXT = 72;    // worst-case number of M tiles

// ws header (int offsets)
constexpr int HDR_CNT = 0;    // [8]  per-expert routed count
constexpr int HDR_NT  = 8;    // [1]  total M tiles
constexpr int HDR_TE  = 16;   // [96] tile -> expert
constexpr int HDR_TL  = 112;  // [96] tile -> local row in expert list
constexpr int HDR_TG  = 208;  // [96] tile -> global activation row

__device__ __forceinline__ unsigned short f2bf(float f) {
  unsigned int u = __float_as_uint(f);
  u += 0x7FFFu + ((u >> 16) & 1u);          // round-to-nearest-even
  return (unsigned short)(u >> 16);
}

// Load one 16x32 bf16 A/B fragment from an LDS tile (row-major, stride LSTR).
// Per CDNA5 ISA: lanes 0-15 hold row (lane&15) with K={0..7,16..23},
// lanes 16-31 hold K={8..15,24..31}.
__device__ __forceinline__ v16bf frag_ld(const unsigned short* tile, int r, int half) {
  const unsigned short* p = tile + r * LSTR + half * 8;
  v16bf f;
  ((uint4*)&f)[0] = *(const uint4*)(p);        // K = half*8 .. half*8+7
  ((uint4*)&f)[1] = *(const uint4*)(p + 16);   // K = 16+half*8 ..
  return f;
}

__device__ __forceinline__ v8f wmma_bf16(v16bf a, v16bf b, v8f c) {
  return __builtin_amdgcn_wmma_f32_16x16x32_bf16(false, a, false, b, (short)0, c,
                                                 false, false);
}

// ---------------------------------------------------------------------------
__global__ __launch_bounds__(256)
void moe_zero_init(float* __restrict__ out, int* __restrict__ hdr, size_t n) {
  size_t i = (size_t)blockIdx.x * blockDim.x + threadIdx.x;
  if (i < 16) hdr[i] = 0;
  for (; i < n; i += (size_t)gridDim.x * blockDim.x) out[i] = 0.0f;
}

__global__ __launch_bounds__(256)
void moe_convert_x(const float* __restrict__ x, unsigned short* __restrict__ xb, size_t n) {
  size_t i = (size_t)blockIdx.x * blockDim.x + threadIdx.x;
  for (; i < n; i += (size_t)gridDim.x * blockDim.x) xb[i] = f2bf(x[i]);
}

__global__ __launch_bounds__(256)
void moe_route_build(const int* __restrict__ sel, const float* __restrict__ rw,
                     int* __restrict__ hdr, int* __restrict__ tokl,
                     float* __restrict__ wl) {
  int t = blockIdx.x * blockDim.x + threadIdx.x;
  if (t >= NTOK) return;
  int   e0 = sel[t * 2 + 0], e1 = sel[t * 2 + 1];
  float r0 = rw[t * 2 + 0],  r1 = rw[t * 2 + 1];
  if (e0 == e1) {
    int p = atomicAdd(&hdr[HDR_CNT + e0], 1);
    tokl[e0 * NTOK + p] = t; wl[e0 * NTOK + p] = r0 + r1;
  } else {
    int p0 = atomicAdd(&hdr[HDR_CNT + e0], 1);
    tokl[e0 * NTOK + p0] = t; wl[e0 * NTOK + p0] = r0;
    int p1 = atomicAdd(&hdr[HDR_CNT + e1], 1);
    tokl[e1 * NTOK + p1] = t; wl[e1 * NTOK + p1] = r1;
  }
}

__global__ __launch_bounds__(64)
void moe_route_finalize(int* __restrict__ hdr, int* __restrict__ tokl,
                        float* __restrict__ wl) {
  int tid = threadIdx.x;
  if (tid < NE) {   // pad each expert segment to a multiple of TM (weight 0)
    int c  = hdr[HDR_CNT + tid];
    int pc = (c + TM - 1) & ~(TM - 1);
    for (int i = c; i < pc; ++i) { tokl[tid * NTOK + i] = 0; wl[tid * NTOK + i] = 0.0f; }
  }
  __syncthreads();
  if (tid == 0) {   // build tile map (<= 71 tiles)
    int rows = 0, nt = 0;
    for (int e = 0; e < NE; ++e) {
      int c  = hdr[HDR_CNT + e];
      int pc = (c + TM - 1) & ~(TM - 1);
      for (int j = 0; j < pc / TM; ++j) {
        hdr[HDR_TE + nt] = e;
        hdr[HDR_TL + nt] = j * TM;
        hdr[HDR_TG + nt] = rows + j * TM;
        ++nt;
      }
      rows += pc;
    }
    hdr[HDR_NT] = nt;
  }
}

// ---------------------------------------------------------------------------
// GEMM1: act[row, n] = silu(x[row,:] @ Wg[n,:]) * (x[row,:] @ Wu[n,:])
// A: gathered bf16 tokens (irregular rows -> VGPR staging); B: fp32 weights
// converted to bf16 while staging (conversion needs VALU, so no TDM here).
// ---------------------------------------------------------------------------
__global__ __launch_bounds__(256)
void moe_gemm_gateup(const unsigned short* __restrict__ xb,
                     const float* __restrict__ gup,
                     const int* __restrict__ hdr,
                     const int* __restrict__ tokl,
                     unsigned short* __restrict__ act) {
  const int nt = hdr[HDR_NT];
  const int mt = blockIdx.x;
  if (mt >= nt) return;
  const int e    = hdr[HDR_TE + mt];
  const int lrow = hdr[HDR_TL + mt];
  const int grow = hdr[HDR_TG + mt];
  const int n0   = blockIdx.y * TN;

  __shared__ unsigned short lsA [TM * LSTR];
  __shared__ unsigned short lsBg[TN * LSTR];
  __shared__ unsigned short lsBu[TN * LSTR];
  __shared__ int s_tok[TM];

  const int tid  = threadIdx.x, lane = tid & 31, wave = tid >> 5;
  const int wm   = wave & 3, wn = wave >> 2;        // 4x2 wave grid, 32x32 each
  const int r    = lane & 15, half = lane >> 4;

  if (tid < TM) s_tok[tid] = tokl[e * NTOK + lrow + tid];
  __syncthreads();

  const float* Wg = gup + ((size_t)e * (2 * INTR) + n0) * HID;
  const float* Wu = Wg + (size_t)INTR * HID;

  v8f cg[2][2] = {};
  v8f cu[2][2] = {};

  for (int k0 = 0; k0 < HID; k0 += TK) {
    // ---- stage A: 128x32 bf16 (gathered rows), 2 x 16B per thread
#pragma unroll
    for (int i = 0; i < 2; ++i) {
      int idx = tid + i * 256;                 // 0..511
      int row = idx >> 2, c8 = (idx & 3) * 8;
      *(uint4*)&lsA[row * LSTR + c8] =
          *(const uint4*)(xb + (size_t)s_tok[row] * HID + k0 + c8);
    }
    // ---- stage Bgate/Bup: 64x32 fp32 -> bf16 each
#pragma unroll
    for (int i = 0; i < 2; ++i) {
      int idx = tid + i * 256;
      int row = idx >> 3, c4 = (idx & 7) * 4;
      float4 g = *(const float4*)(Wg + (size_t)row * HID + k0 + c4);
      float4 u = *(const float4*)(Wu + (size_t)row * HID + k0 + c4);
      uint2 pg, pu;
      pg.x = (unsigned)f2bf(g.x) | ((unsigned)f2bf(g.y) << 16);
      pg.y = (unsigned)f2bf(g.z) | ((unsigned)f2bf(g.w) << 16);
      pu.x = (unsigned)f2bf(u.x) | ((unsigned)f2bf(u.y) << 16);
      pu.y = (unsigned)f2bf(u.z) | ((unsigned)f2bf(u.w) << 16);
      *(uint2*)&lsBg[row * LSTR + c4] = pg;
      *(uint2*)&lsBu[row * LSTR + c4] = pu;
    }
    if (k0 + TK < HID) {   // global_prefetch_b8 of next weight lines
      __builtin_prefetch(Wg + (size_t)(tid >> 2) * HID + k0 + TK, 0, 1);
      __builtin_prefetch(Wu + (size_t)(tid >> 2) * HID + k0 + TK, 0, 1);
    }
    __syncthreads();

    v16bf a0 = frag_ld(&lsA [(wm * 32 +  0) * LSTR], r, half);
    v16bf a1 = frag_ld(&lsA [(wm * 32 + 16) * LSTR], r, half);
    v16bf g0 = frag_ld(&lsBg[(wn * 32 +  0) * LSTR], r, half);
    v16bf g1 = frag_ld(&lsBg[(wn * 32 + 16) * LSTR], r, half);
    v16bf u0 = frag_ld(&lsBu[(wn * 32 +  0) * LSTR], r, half);
    v16bf u1 = frag_ld(&lsBu[(wn * 32 + 16) * LSTR], r, half);

    cg[0][0] = wmma_bf16(a0, g0, cg[0][0]);
    cg[0][1] = wmma_bf16(a0, g1, cg[0][1]);
    cg[1][0] = wmma_bf16(a1, g0, cg[1][0]);
    cg[1][1] = wmma_bf16(a1, g1, cg[1][1]);
    cu[0][0] = wmma_bf16(a0, u0, cu[0][0]);
    cu[0][1] = wmma_bf16(a0, u1, cu[0][1]);
    cu[1][0] = wmma_bf16(a1, u0, cu[1][0]);
    cu[1][1] = wmma_bf16(a1, u1, cu[1][1]);
    __syncthreads();
  }

  // ---- epilogue: silu(gate) * up -> bf16 activations
#pragma unroll
  for (int mi = 0; mi < 2; ++mi)
#pragma unroll
    for (int ni = 0; ni < 2; ++ni)
#pragma unroll
      for (int v = 0; v < 8; ++v) {
        int M = wm * 32 + mi * 16 + v + half * 8;        // C layout: lanes16-31 -> M+8
        int N = n0 + wn * 32 + ni * 16 + r;
        float g = cg[mi][ni][v], u = cu[mi][ni][v];
        float s = g / (1.0f + __expf(-g));
        act[(size_t)(grow + M) * INTR + N] = f2bf(s * u);
      }
}

// ---------------------------------------------------------------------------
// GEMM2: out[tok, h] += w_row * (act[row,:] @ Wd[h,:])
// A tile (128x32 bf16, regular stride) is DMA'd into LDS by the Tensor Data
// Mover; D# pad fields (interval=16 DW, amount=8 DW) reproduce the 96 B LDS
// row stride so the fragment loader is shared with the VGPR-staged path.
// ---------------------------------------------------------------------------
__global__ __launch_bounds__(256)
void moe_gemm_down(const unsigned short* __restrict__ act,
                   const float* __restrict__ dwn,
                   const int* __restrict__ hdr,
                   const int* __restrict__ tokl,
                   const float* __restrict__ wl,
                   float* __restrict__ out) {
  const int nt = hdr[HDR_NT];
  const int mt = blockIdx.x;
  if (mt >= nt) return;
  const int e    = hdr[HDR_TE + mt];
  const int lrow = hdr[HDR_TL + mt];
  const int grow = hdr[HDR_TG + mt];
  const int h0   = blockIdx.y * TN;

  __shared__ unsigned short lsA[TM * LSTR];
  __shared__ unsigned short lsB[TN * LSTR];

  const int tid  = threadIdx.x, lane = tid & 31, wave = tid >> 5;
  const int wm   = wave & 3, wn = wave >> 2;
  const int r    = lane & 15, half = lane >> 4;

  const float* Wd = dwn + ((size_t)e * HID + h0) * INTR;
  const unsigned short* ab = act + (size_t)grow * INTR;

  // --- Tensor DMA descriptor (D#), CDNA5 ISA ---------------
  // group1: workgroup_mask=0 | data_size=1(2B) | pad_enable | pad_interval=3
  // (16 DWORDs = one 64B tile row) | pad_amount=7 (8 DWORDs = 32B pad)
  // tensor_dim0=32, tensor_dim1=128, tile_dim0=32, tile_dim1=128,
  // tensor_dim0_stride=INTR elements; 2-D tile -> groups 2/3 zero.
  const v8i g1 = { (1 << 16) | (1 << 20) | (3 << 22) | (7 << 25),
                   (TK << 16),                 // tensor_dim0[15:0] in [31:16]
                   (TM << 16),                 // dim0 hi16=0 | tensor_dim1 lo16
                   (TK << 16),                 // dim1 hi16=0 | tile_dim0
                   TM,                         // tile_dim1 | tile_dim2=0
                   INTR,                       // tensor_dim0_stride lo32
                   0, 0 };                     // stride0 hi | stride1
  const v4i gz4 = { 0, 0, 0, 0 };
  const v8i gz8 = { 0, 0, 0, 0, 0, 0, 0, 0 };
  const unsigned lds_a = (unsigned)(uintptr_t)(&lsA[0]);   // LDS byte offset

  v8f c[2][2] = {};

  for (int k0 = 0; k0 < INTR; k0 += TK) {
    // ---- A tile: one TDM transfer per K-step, issued by wave 0 only
    if (wave == 0) {
      uint64_t ga = (uint64_t)(uintptr_t)(ab + k0);        // byte address
      v4u g0 = { 1u,                                       // count=1, user mode
                 lds_a,                                    // lds_addr
                 (unsigned)(ga & 0xFFFFFFFFu),             // global_addr[31:0]
                 (unsigned)((ga >> 32) & 0x1FFFFFFu) | (2u << 30) };  // [56:32]|type=2
      // 6-arg toolchain form: (g0, g1, g2, g3, extra_v8i, cpol)
      __builtin_amdgcn_tensor_load_to_lds(g0, g1, gz4, gz4, gz8, 0);
    }
    // ---- stage B: 64x32 fp32 -> bf16 (overlaps with the TDM transfer)
#pragma unroll
    for (int i = 0; i < 2; ++i) {
      int idx = tid + i * 256;
      int row = idx >> 3, c4 = (idx & 7) * 4;
      float4 d = *(const float4*)(Wd + (size_t)row * INTR + k0 + c4);
      uint2 pd;
      pd.x = (unsigned)f2bf(d.x) | ((unsigned)f2bf(d.y) << 16);
      pd.y = (unsigned)f2bf(d.z) | ((unsigned)f2bf(d.w) << 16);
      *(uint2*)&lsB[row * LSTR + c4] = pd;
    }
    if (k0 + TK < INTR)
      __builtin_prefetch(Wd + (size_t)(tid >> 2) * INTR + k0 + TK, 0, 1);
    if (wave == 0) __builtin_amdgcn_s_wait_tensorcnt(0);
    __syncthreads();

    v16bf a0 = frag_ld(&lsA[(wm * 32 +  0) * LSTR], r, half);
    v16bf a1 = frag_ld(&lsA[(wm * 32 + 16) * LSTR], r, half);
    v16bf b0 = frag_ld(&lsB[(wn * 32 +  0) * LSTR], r, half);
    v16bf b1 = frag_ld(&lsB[(wn * 32 + 16) * LSTR], r, half);

    c[0][0] = wmma_bf16(a0, b0, c[0][0]);
    c[0][1] = wmma_bf16(a0, b1, c[0][1]);
    c[1][0] = wmma_bf16(a1, b0, c[1][0]);
    c[1][1] = wmma_bf16(a1, b1, c[1][1]);
    __syncthreads();
  }

  // ---- epilogue: weighted scatter (<=2 commutative fp32 adds per out elem)
#pragma unroll
  for (int mi = 0; mi < 2; ++mi)
#pragma unroll
    for (int ni = 0; ni < 2; ++ni)
#pragma unroll
      for (int v = 0; v < 8; ++v) {
        int M = wm * 32 + mi * 16 + v + half * 8;
        int N = h0 + wn * 32 + ni * 16 + r;
        int li = e * NTOK + lrow + M;
        int   tok = tokl[li];
        float w   = wl[li];
        atomicAdd(&out[(size_t)tok * HID + N], c[mi][ni][v] * w);
      }
}

// ---------------------------------------------------------------------------
extern "C" void kernel_launch(void* const* d_in, const int* in_sizes, int n_in,
                              void* d_out, int out_size, void* d_ws, size_t ws_size,
                              hipStream_t stream) {
  (void)in_sizes; (void)n_in; (void)out_size; (void)ws_size;
  const float* x    = (const float*)d_in[0];   // [T, H]
  const int*   sel  = (const int*)  d_in[1];   // [T, 2]
  const float* rw   = (const float*)d_in[2];   // [T, 2]
  const float* gup  = (const float*)d_in[3];   // [E, 2I, H]
  const float* dwn  = (const float*)d_in[4];   // [E, H, I]
  float*       out  = (float*)d_out;           // [T, H]

  char* ws = (char*)d_ws;
  int*            hdr  = (int*)ws;                                   // 4 KB header
  int*            tokl = (int*)(ws + 4096);                          // E*T ints
  float*          wl   = (float*)(ws + 4096 + (size_t)NE * NTOK * 4);
  unsigned short* xb   = (unsigned short*)(ws + 4096 + 2 * (size_t)NE * NTOK * 4);
  unsigned short* act  = xb + (size_t)NTOK * HID;                    // <= 9216 rows x I

  const size_t nout = (size_t)NTOK * HID;
  moe_zero_init    <<<2048, 256, 0, stream>>>(out, hdr, nout);
  moe_convert_x    <<<2048, 256, 0, stream>>>(x, xb, nout);
  moe_route_build  <<<NTOK / 256, 256, 0, stream>>>(sel, rw, hdr, tokl, wl);
  moe_route_finalize<<<1, 64, 0, stream>>>(hdr, tokl, wl);
  moe_gemm_gateup  <<<dim3(MAXT, INTR / TN), 256, 0, stream>>>(xb, gup, hdr, tokl, act);
  moe_gemm_down    <<<dim3(MAXT, HID  / TN), 256, 0, stream>>>(act, dwn, hdr, tokl, wl, out);
}
